// LSTMLayer_57887569215830
// MI455X (gfx1250) — matrix operand order repaired and checked by reading, not compiled, not run on hardware
//
#include <hip/hip_runtime.h>
#include <hip/hip_bf16.h>

// ---------------------------------------------------------------------------
// LSTM layer, persistent-kernel design for MI455X (gfx1250, wave32, WMMA+TDM).
// B=32, D=256, T=1024, H=512.  16 WGPs, each owns 32 hidden units.
// Per step, per WGP:  g[32x128] = [x_t | h_prev][32x768] * Wslice^T[768x128]
// via v_wmma_f32_16x16x32_bf16.  Weights live in LDS for the whole kernel.
// x_{t+1} is prefetched into a double-buffered LDS region by the Tensor Data
// Mover (tensor_load_to_lds, TENSORcnt) while step t computes.
// ---------------------------------------------------------------------------

typedef __attribute__((ext_vector_type(16))) __bf16 v16bf;
typedef __attribute__((ext_vector_type(8)))  __bf16 v8bf;
typedef __attribute__((ext_vector_type(8)))  float  v8f;
typedef __attribute__((ext_vector_type(4)))  unsigned int tdm_v4u;
typedef __attribute__((ext_vector_type(8)))  int          tdm_v8i;
typedef __attribute__((ext_vector_type(4)))  int          tdm_v4i;

#define BB   32      // batch
#define DD   256     // input dim
#define HH   512     // hidden dim
#define TT   1024    // timesteps
#define NWG  16      // workgroups (one per WGP)
#define HSL  32      // hidden units per WGP
#define NSL  128     // gate rows per WGP (4 * HSL)
#define KK   768     // fused K = DD + HH
#define SWP  776     // W slice LDS stride (bf16): 1552B -> rows shift 4 banks
#define SXP  264     // x tile LDS stride (bf16): 132 dwords -> rows shift 4 banks (TDM pad)
#define SHP  520     // h tile LDS stride (bf16): 260 dwords -> rows shift 4 banks
#define GBP  132     // padded f32 stride for gate buffer

// ---- workspace layout (bytes) ----
#define WIH_OFF  0ull                               // 4H*D bf16  = 1 MB
#define WHH_OFF  (WIH_OFF + 2048ull*256*2)          // 4H*H bf16  = 2 MB
#define BS_OFF   (WHH_OFF + 2048ull*512*2)          // 4H f32     = 8 KB
#define XT_OFF   (BS_OFF  + 2048ull*4)              // T*B*D bf16 = 16 MB
#define HB_OFF   (XT_OFF  + (size_t)TT*BB*DD*2)     // 2 * B*H bf16 (double buffer)
#define CTR_OFF  (HB_OFF  + 2ull*BB*HH*2)           // barrier counter

__device__ __forceinline__ float sigmoid_f(float x) {
    return 1.0f / (1.0f + __expf(-x));
}
__device__ __forceinline__ float tanh_f(float x) {
    x = fminf(15.0f, fmaxf(-15.0f, x));
    float e = __expf(2.0f * x);
    return (e - 1.0f) / (e + 1.0f);
}
__device__ __forceinline__ v16bf cat16(v8bf lo, v8bf hi) {
    return __builtin_shufflevector(lo, hi, 0,1,2,3,4,5,6,7,8,9,10,11,12,13,14,15);
}

// Issue a TDM load of one x_t slab [32 rows x 256 bf16, contiguous] into LDS
// at byte offset lds_off, with LDS padding of 4 DWORDs every 128 DWORDs so the
// destination rows land on the bank-conflict-free SXP=264-element stride.
__device__ __forceinline__ void tdm_load_x(const __bf16* gsrc, unsigned lds_off) {
    unsigned long long ga = (unsigned long long)(uintptr_t)gsrc;
    tdm_v4u g0;
    g0[0] = 1u;                                    // count=1, user descriptor
    g0[1] = lds_off;                               // lds_addr
    g0[2] = (unsigned)(ga & 0xffffffffu);          // global_addr[31:0]
    g0[3] = (unsigned)((ga >> 32) & 0x01ffffffu)   // global_addr[56:32]
          | (2u << 30);                            // type = 2 ("image")
    tdm_v8i g1;
    g1[0] = (int)((1u << 16)      // data_size = 1 -> 2 bytes
                | (1u << 20)      // pad_enable
                | (6u << 22)      // pad_interval: code 6 -> every 128 DWORDs
                | (3u << 25));    // pad_amount:   code 3 -> 4 DWORDs
    g1[1] = (int)(256u << 16);    // [63:48]  tensor_dim0 lo16 = 256
    g1[2] = (int)(32u  << 16);    // [79:64] dim0 hi=0, [95:80] tensor_dim1 lo16 = 32
    g1[3] = (int)(256u << 16);    // [111:96] dim1 hi=0, [127:112] tile_dim0 = 256
    g1[4] = (int)32;              // [143:128] tile_dim1 = 32, [159:144] tile_dim2 = 0
    g1[5] = (int)256;             // [191:160] tensor_dim0_stride lo32 = 256
    g1[6] = 0;
    g1[7] = 0;
    tdm_v4i z4 = {0, 0, 0, 0};
    tdm_v8i z8 = {0, 0, 0, 0, 0, 0, 0, 0};
    __builtin_amdgcn_tensor_load_to_lds(g0, g1, z4, z4, z8, 0);
}

// ---------------------------------------------------------------------------
// Kernel 1: convert weights to bf16, sum biases, zero h buffers + barrier ctr.
// ---------------------------------------------------------------------------
__global__ void lstm_setup(const float* __restrict__ W_ih, const float* __restrict__ W_hh,
                           const float* __restrict__ b_ih, const float* __restrict__ b_hh,
                           __bf16* __restrict__ wih, __bf16* __restrict__ whh,
                           float* __restrict__ bsum, __bf16* __restrict__ hbuf,
                           unsigned* __restrict__ ctr) {
    size_t i = (size_t)blockIdx.x * blockDim.x + threadIdx.x;
    if (i < 2048ull * 256) wih[i] = (__bf16)W_ih[i];
    if (i < 2048ull * 512) whh[i] = (__bf16)W_hh[i];
    if (i < 2048)          bsum[i] = b_ih[i] + b_hh[i];
    if (i < 2ull * BB * HH) hbuf[i] = (__bf16)0.0f;
    if (i == 0)            *ctr = 0u;
}

// ---------------------------------------------------------------------------
// Kernel 2: transpose x [B,D,T] f32 -> xt [T,B,D] bf16 (coalesced both ways).
// ---------------------------------------------------------------------------
__global__ void lstm_xpose(const float* __restrict__ x, __bf16* __restrict__ xt) {
    __shared__ float tile[32][33];
    const int b  = blockIdx.x;
    const int d0 = blockIdx.y * 32;
    const int t0 = blockIdx.z * 32;
    const int j  = threadIdx.x & 31;
    const int i0 = threadIdx.x >> 5;
#pragma unroll
    for (int q = 0; q < 4; ++q) {
        int i = i0 + q * 8;
        tile[i][j] = x[((size_t)(b * DD + d0 + i)) * TT + t0 + j];
    }
    __syncthreads();
#pragma unroll
    for (int q = 0; q < 4; ++q) {
        int i = i0 + q * 8;  // t-offset
        xt[((size_t)(t0 + i)) * (BB * DD) + (size_t)b * DD + d0 + j] = (__bf16)tile[j][i];
    }
}

// ---------------------------------------------------------------------------
// Kernel 3: persistent LSTM recurrence. Grid = 16 blocks x 256 threads.
// ---------------------------------------------------------------------------
__global__ __launch_bounds__(256, 1) void lstm_persistent(
    const __bf16* __restrict__ wih, const __bf16* __restrict__ whh,
    const float*  __restrict__ bsum, const __bf16* __restrict__ xt,
    __bf16* __restrict__ hbuf, unsigned* __restrict__ ctr,
    float* __restrict__ out) {

    __shared__ __bf16 sW[NSL * SWP];       // weight slice [128 x 768] (padded)
    __shared__ __bf16 sX[2 * BB * SXP];    // x_t double buffer (TDM-filled)
    __shared__ __bf16 sH[BB * SHP];        // h_prev staging [32 x 512] (padded)
    __shared__ float  gbuf[BB * GBP];      // gate pre-activations [32 x 128]
    __shared__ float  sbias[NSL];

    const int tid   = threadIdx.x;
    const int wg    = blockIdx.x;          // 0..15
    const int hbase = wg * HSL;
    const int lane  = tid & 31;
    const int w     = tid >> 5;            // wave 0..7
    const int lhi   = lane >> 4;           // 0/1 half-wave
    const int lm    = lane & 15;

    // ---- one-time: load weight slice + bias into LDS ----
    // LDS row r (0..127) <-> global gate row gr = (r>>5)*H + hbase + (r&31)
    for (int ch = tid; ch < NSL * (KK / 8); ch += 256) {   // 12288 uint4 chunks
        int r  = ch / (KK / 8);
        int cc = ch % (KK / 8);                            // 8-elem chunk in 0..95
        int gr = (r >> 5) * HH + hbase + (r & 31);
        uint4 v;
        if (cc < DD / 8) v = ((const uint4*)(wih + (size_t)gr * DD))[cc];
        else             v = ((const uint4*)(whh + (size_t)gr * HH))[cc - DD / 8];
        *(uint4*)(sW + r * SWP + cc * 8) = v;
    }
    if (tid < NSL) sbias[tid] = bsum[(tid >> 5) * HH + hbase + (tid & 31)];

    // ---- prologue: TDM-load x_0 into sX buffer 0 (wave 0 issues) ----
    const unsigned sx_base = (unsigned)(uintptr_t)(void*)sX;
    if (w == 0) tdm_load_x(xt, sx_base);

    float creg[4] = {0.f, 0.f, 0.f, 0.f};  // cell state: 1024 units / 256 threads

    const int nb = w * 16;                 // this wave's N columns [nb, nb+16)
    const __bf16* brow = sW + (nb + lm) * SWP;
    const __bf16* ah0  = sH + lm * SHP;
    const __bf16* ah1  = sH + (16 + lm) * SHP;

    for (int t = 0; t < TT; ++t) {
        const __bf16* hprev = hbuf + (size_t)(t & 1) * BB * HH;
        __bf16*       hnext = hbuf + (size_t)((t + 1) & 1) * BB * HH;

        // ---- stage h_prev into LDS (uint4 chunks, padded stride) ----
        for (int ch = tid; ch < BB * HH / 8; ch += 256) {      // 2048 chunks
            int r = ch >> 6, c = (ch & 63) * 8;
            *(uint4*)(sH + r * SHP + c) = ((const uint4*)hprev)[ch];
        }
        // ---- wave 0: wait for x_t DMA, kick off x_{t+1} DMA ----
        if (w == 0) {
            __builtin_amdgcn_s_wait_tensorcnt(0);
            if (t + 1 < TT)
                tdm_load_x(xt + (size_t)(t + 1) * BB * DD,
                           sx_base + (unsigned)(((t + 1) & 1) * BB * SXP * 2));
        }
        __syncthreads();

        // ---- WMMA: g = [x_t | h] * Wslice^T, two 16x16 M-tiles per wave ----
        v8f c0 = {}; v8f c1 = {};
        const __bf16* sXb = sX + (t & 1) * (BB * SXP);
        const __bf16* ax0 = sXb + lm * SXP;
        const __bf16* ax1 = sXb + (16 + lm) * SXP;
#pragma unroll
        for (int k0 = 0; k0 < DD; k0 += 32) {                  // x contribution
            v16bf bf = cat16(*(const v8bf*)(brow + k0 + lhi * 8),
                             *(const v8bf*)(brow + k0 + 16 + lhi * 8));
            v16bf a0 = cat16(*(const v8bf*)(ax0 + k0 + lhi * 8),
                             *(const v8bf*)(ax0 + k0 + 16 + lhi * 8));
            v16bf a1 = cat16(*(const v8bf*)(ax1 + k0 + lhi * 8),
                             *(const v8bf*)(ax1 + k0 + 16 + lhi * 8));
            c0 = __builtin_amdgcn_wmma_f32_16x16x32_bf16(false, a0, false, bf,
                                                         (short)0, c0, false, false);
            c1 = __builtin_amdgcn_wmma_f32_16x16x32_bf16(false, a1, false, bf,
                                                         (short)0, c1, false, false);
        }
#pragma unroll 4
        for (int k0 = 0; k0 < HH; k0 += 32) {                  // h contribution
            v16bf bf = cat16(*(const v8bf*)(brow + DD + k0 + lhi * 8),
                             *(const v8bf*)(brow + DD + k0 + 16 + lhi * 8));
            v16bf a0 = cat16(*(const v8bf*)(ah0 + k0 + lhi * 8),
                             *(const v8bf*)(ah0 + k0 + 16 + lhi * 8));
            v16bf a1 = cat16(*(const v8bf*)(ah1 + k0 + lhi * 8),
                             *(const v8bf*)(ah1 + k0 + 16 + lhi * 8));
            c0 = __builtin_amdgcn_wmma_f32_16x16x32_bf16(false, a0, false, bf,
                                                         (short)0, c0, false, false);
            c1 = __builtin_amdgcn_wmma_f32_16x16x32_bf16(false, a1, false, bf,
                                                         (short)0, c1, false, false);
        }
        // C/D layout: VGPR v -> (M = v + 8*lhi, N = lm)
#pragma unroll
        for (int v = 0; v < 8; ++v) {
            gbuf[(v + 8 * lhi) * GBP + nb + lm]      = c0[v];
            gbuf[(16 + v + 8 * lhi) * GBP + nb + lm] = c1[v];
        }
        __syncthreads();

        // ---- gates: i,f,g,o -> update c (registers), emit h ----
#pragma unroll
        for (int q = 0; q < 4; ++q) {
            int idx = tid + q * 256;
            int b = idx >> 5, j = idx & 31;
            float gi = gbuf[b * GBP + j]      + sbias[j];
            float gf = gbuf[b * GBP + 32 + j] + sbias[32 + j];
            float gg = gbuf[b * GBP + 64 + j] + sbias[64 + j];
            float go = gbuf[b * GBP + 96 + j] + sbias[96 + j];
            float iv = sigmoid_f(gi);
            float fv = sigmoid_f(gf);
            float gv = tanh_f(gg);
            float ov = sigmoid_f(go);
            float c  = fv * creg[q] + iv * gv;
            creg[q]  = c;
            float h  = ov * tanh_f(c);
            out[((size_t)b * HH + hbase + j) * TT + t] = h;    // [B,H,T]
            hnext[b * HH + hbase + j] = (__bf16)h;
        }

        // ---- grid barrier (monotonic counter, double-buffered h) ----
        __threadfence();                       // release h stores device-wide
        __syncthreads();
        if (tid == 0) {
            atomicAdd(ctr, 1u);
            unsigned tgt = (unsigned)NWG * (unsigned)(t + 1);
            while (__hip_atomic_load(ctr, __ATOMIC_RELAXED,
                                     __HIP_MEMORY_SCOPE_AGENT) < tgt)
                __builtin_amdgcn_s_sleep(1);
        }
        __syncthreads();
        __threadfence();                       // acquire before reading hnext
    }
}

// ---------------------------------------------------------------------------
extern "C" void kernel_launch(void* const* d_in, const int* in_sizes, int n_in,
                              void* d_out, int out_size, void* d_ws, size_t ws_size,
                              hipStream_t stream) {
    const float* x    = (const float*)d_in[0];   // [B, D, T]
    const float* W_ih = (const float*)d_in[1];   // [4H, D]
    const float* W_hh = (const float*)d_in[2];   // [4H, H]
    const float* b_ih = (const float*)d_in[3];   // [4H]
    const float* b_hh = (const float*)d_in[4];   // [4H]
    float* out = (float*)d_out;                  // [B, H, T]

    char* ws = (char*)d_ws;
    __bf16*   wih  = (__bf16*)(ws + WIH_OFF);
    __bf16*   whh  = (__bf16*)(ws + WHH_OFF);
    float*    bsum = (float*)(ws + BS_OFF);
    __bf16*   xt   = (__bf16*)(ws + XT_OFF);
    __bf16*   hbuf = (__bf16*)(ws + HB_OFF);
    unsigned* ctr  = (unsigned*)(ws + CTR_OFF);

    // 1) weight/bias conversion + state zeroing (covers graph replays)
    lstm_setup<<<4096, 256, 0, stream>>>(W_ih, W_hh, b_ih, b_hh,
                                         wih, whh, bsum, hbuf, ctr);
    // 2) x -> [T,B,D] bf16
    lstm_xpose<<<dim3(BB, DD / 32, TT / 32), 256, 0, stream>>>(x, xt);
    // 3) persistent recurrence: 16 WGPs, weights LDS-resident, TDM x-prefetch
    lstm_persistent<<<NWG, 256, 0, stream>>>(wih, whh, bsum, xt, hbuf, ctr, out);
}
